// StructuredSlotCommunication_83562883711091
// MI455X (gfx1250) — matrix operand order repaired and checked by reading, not compiled, not run on hardware
//
#include <hip/hip_runtime.h>

// Problem constants
#define BB  32768
#define SSQ 3
#define HH  1024
#define NHH 16
#define HDD 64

// Tiling
#define NB     16            // batches per workgroup
#define MROWS  48            // NB * 3 rows, == 3 WMMA m-tiles of 16
#define SAW    1032          // bf16 row stride for sA/sX and sCtx (pad 8)
#define SQW    200           // bf16 row stride for per-head qkv staging (192 + 8)

// Dynamic LDS layout (bytes)
#define OFF_A      0                         // 48*1032*2 = 99072
#define OFF_CTX    99072                     // 48*1032*2 = 99072
#define OFF_Q      198144                    // 48*200*2  = 19200
#define OFF_S      217344                    // 16*12*4   = 768
#define SMEM_BYTES 218112

typedef __attribute__((ext_vector_type(16))) __bf16 v16bf;
typedef __attribute__((ext_vector_type(8)))  __bf16 v8bf;
typedef __attribute__((ext_vector_type(8)))  float  v8f;

__device__ __forceinline__ __bf16 f2bf(float f) { return (__bf16)f; }

// A-fragment (16x32 bf16, MxK) from LDS, contiguous-k layout (same as B):
//  lane<16 : M=lane,    elems 0..15 -> K=k0+0..15
//  lane>=16: M=lane-16, elems 0..15 -> K=k0+16..31
__device__ __forceinline__ v16bf load_a_frag(const __bf16* base, int lane, int m0, int k0) {
  int m   = m0 + (lane & 15);
  int off = (lane >> 4) * 16;
  const __bf16* p = base + m * SAW + k0 + off;
  v8bf lo = *(const v8bf*)(p);
  v8bf hi = *(const v8bf*)(p + 8);
  return __builtin_shufflevector(lo, hi, 0,1,2,3,4,5,6,7,8,9,10,11,12,13,14,15);
}

// B-fragment (32x16 bf16, KxN) from row-major f32 weight W[n][k] (ldw=K stride):
//  lane&15 = N column; lane<16 -> K=k0+0..15, lane>=16 -> K=k0+16..31
// Also prefetches next K-iteration's 64B chunk (weights are L2-resident).
__device__ __forceinline__ v16bf load_b_frag_f32(const float* __restrict__ w, int ldw,
                                                 int lane, int n0, int k0) {
  int n    = n0 + (lane & 15);
  int koff = (lane >> 4) * 16;
  const float* p = w + (long)n * ldw + k0 + koff;
  const float4* p4 = (const float4*)p;
  float4 f0 = p4[0], f1 = p4[1], f2 = p4[2], f3 = p4[3];
  __builtin_prefetch(p + 32, 0, 1);   // next k0 chunk -> global_prefetch_b8
  v16bf b;
  b[0]=f2bf(f0.x);  b[1]=f2bf(f0.y);  b[2]=f2bf(f0.z);  b[3]=f2bf(f0.w);
  b[4]=f2bf(f1.x);  b[5]=f2bf(f1.y);  b[6]=f2bf(f1.z);  b[7]=f2bf(f1.w);
  b[8]=f2bf(f2.x);  b[9]=f2bf(f2.y);  b[10]=f2bf(f2.z); b[11]=f2bf(f2.w);
  b[12]=f2bf(f3.x); b[13]=f2bf(f3.y); b[14]=f2bf(f3.z); b[15]=f2bf(f3.w);
  return b;
}

#define WMMA_BF16(A, Bm, C) \
  __builtin_amdgcn_wmma_f32_16x16x32_bf16(false, (A), false, (Bm), (short)0, (C), false, false)

__global__ __launch_bounds__(256)
void StructuredSlotCommunication_83562883711091_kernel(
    const float* __restrict__ slots, const float* __restrict__ w1,
    const float* __restrict__ b1,    const float* __restrict__ w2,
    const float* __restrict__ b2,    const float* __restrict__ gamma,
    const float* __restrict__ beta,  float* __restrict__ out) {
  extern __shared__ char smem[];
  __bf16* sA   = (__bf16*)(smem + OFF_A);    // slots tile bf16, later reused as sX
  __bf16* sCtx = (__bf16*)(smem + OFF_CTX);  // attention context bf16
  __bf16* sQ   = (__bf16*)(smem + OFF_Q);    // per-head q|k|v staging bf16
  float*  sS   = (float*)(smem + OFF_S);     // 3x3 scores / probs per batch

  const int tid  = threadIdx.x;
  const int lane = tid & 31;
  // Wave index as a provably-scalar (SGPR) value: keeps all tile-selection
  // branches scalar so EXEC remains all-ones around every WMMA.
  const int wave = __builtin_amdgcn_readfirstlane(tid >> 5);
  const int rowBase = blockIdx.x * MROWS;    // 48 consecutive (b,s) rows

  // ---------------- Stage slots tile: global f32 -> LDS bf16 ----------------
  {
    const float4* g = (const float4*)(slots + (long)rowBase * HH);
    #pragma unroll 1
    for (int i = 0; i < 48; ++i) {
      int idx = tid + i * 256;          // 0..12287 float4's
      int row = idx >> 8;               // 256 float4 per 1024-col row
      int c4  = idx & 255;
      float4 v = g[idx];
      union { __bf16 h[4]; uint2 u; } pk;
      pk.h[0] = f2bf(v.x); pk.h[1] = f2bf(v.y);
      pk.h[2] = f2bf(v.z); pk.h[3] = f2bf(v.w);
      *(uint2*)(sA + row * SAW + c4 * 4) = pk.u;
    }
  }
  __syncthreads();

  // ---------------- Per-head: QKV GEMM + masked softmax + ctx ---------------
  const bool twoTiles = (wave < 4);           // scalar condition (tiles 8..11)
  for (int head = 0; head < NHH; ++head) {
    // Phase 1: this head's 192 qkv columns (12 N-tiles of 16) over 8 waves.
    v8f acc[2][3];
    #pragma unroll
    for (int nt = 0; nt < 2; ++nt)
      #pragma unroll
      for (int mt = 0; mt < 3; ++mt) { v8f z = {}; acc[nt][mt] = z; }

    const int t0 = wave;                      // always valid (0..7)
    const int t1 = wave + 8;                  // valid only for waves 0..3
    const int e0Base = (t0 >> 2) * HH + head * HDD + (t0 & 3) * 16;
    const int e1Base = (t1 >> 2) * HH + head * HDD + (t1 & 3) * 16;

    #pragma unroll 1
    for (int k0 = 0; k0 < HH; k0 += 32) {
      v16bf a0 = load_a_frag(sA, lane, 0,  k0);
      v16bf a1 = load_a_frag(sA, lane, 16, k0);
      v16bf a2 = load_a_frag(sA, lane, 32, k0);
      {
        v16bf bf = load_b_frag_f32(w1, HH, lane, e0Base, k0);
        acc[0][0] = WMMA_BF16(a0, bf, acc[0][0]);
        acc[0][1] = WMMA_BF16(a1, bf, acc[0][1]);
        acc[0][2] = WMMA_BF16(a2, bf, acc[0][2]);
      }
      if (twoTiles) {
        v16bf bf = load_b_frag_f32(w1, HH, lane, e1Base, k0);
        acc[1][0] = WMMA_BF16(a0, bf, acc[1][0]);
        acc[1][1] = WMMA_BF16(a1, bf, acc[1][1]);
        acc[1][2] = WMMA_BF16(a2, bf, acc[1][2]);
      }
    }
    // Store qkv tiles (+bias) to sQ: cols 0..63=q, 64..127=k, 128..191=v
    #pragma unroll
    for (int nt = 0; nt < 2; ++nt) {
      if (nt == 1 && !twoTiles) break;
      int tile  = (nt == 0) ? t0 : t1;
      int eBase = (nt == 0) ? e0Base : e1Base;
      float bias = b1[eBase + (lane & 15)];
      int lcol = tile * 16 + (lane & 15);
      #pragma unroll
      for (int mt = 0; mt < 3; ++mt)
        #pragma unroll
        for (int r = 0; r < 8; ++r) {
          int m = mt * 16 + (lane >> 4) * 8 + r;   // C-matrix VGPR layout
          sQ[m * SQW + lcol] = f2bf(acc[nt][mt][r] + bias);
        }
    }
    __syncthreads();

    // Scores: 16 batches x 3x3, masked + scaled by 1/sqrt(64)
    if (tid < 144) {
      int lb = tid / 9, r = tid % 9, qs = r / 3, ks = r % 3;
      const __bf16* qp = sQ + (lb * 3 + qs) * SQW;
      const __bf16* kp = sQ + (lb * 3 + ks) * SQW + 64;
      float s = 0.f;
      #pragma unroll 8
      for (int d = 0; d < HDD; ++d) s += (float)qp[d] * (float)kp[d];
      float mask = ((qs == 0 && ks == 1) || (qs == 1 && ks == 0)) ? -1e9f : 0.f;
      sS[lb * 12 + qs * 3 + ks] = s * 0.125f + mask;
    }
    __syncthreads();
    // Softmax over 3 keys
    if (tid < 48) {
      float* sp = sS + (tid / 3) * 12 + (tid % 3) * 3;
      float s0 = sp[0], s1 = sp[1], s2 = sp[2];
      float mx = fmaxf(s0, fmaxf(s1, s2));
      float e0 = __expf(s0 - mx), e1 = __expf(s1 - mx), e2 = __expf(s2 - mx);
      float inv = 1.f / (e0 + e1 + e2);
      sp[0] = e0 * inv; sp[1] = e1 * inv; sp[2] = e2 * inv;
    }
    __syncthreads();
    // ctx = P @ V -> sCtx columns [head*64, head*64+64)
    #pragma unroll 1
    for (int i = 0; i < 12; ++i) {
      int idx = tid + i * 256;                  // 0..3071
      int lb = idx / 192, r = idx % 192, qs = r >> 6, d = r & 63;
      const float*  sp = sS + lb * 12 + qs * 3;
      const __bf16* vp = sQ + (lb * 3) * SQW + 128 + d;
      float c = sp[0] * (float)vp[0] + sp[1] * (float)vp[SQW] + sp[2] * (float)vp[2 * SQW];
      sCtx[(lb * 3 + qs) * SAW + head * HDD + d] = f2bf(c);
    }
    __syncthreads();   // protect sQ before next head overwrites it
  }

  // ---------------- Output projection + bias + residual ---------------------
  __bf16* sX = sA;     // slots tile is dead now; reuse as x = slots + attn_out
  #pragma unroll 1
  for (int j = 0; j < 8; ++j) {
    int n0 = (wave + j * 8) * 16;              // scalar per wave
    v8f c0 = {}, c1 = {}, c2 = {};
    #pragma unroll 1
    for (int k0 = 0; k0 < HH; k0 += 32) {
      v16bf a0 = load_a_frag(sCtx, lane, 0,  k0);
      v16bf a1 = load_a_frag(sCtx, lane, 16, k0);
      v16bf a2 = load_a_frag(sCtx, lane, 32, k0);
      v16bf bf = load_b_frag_f32(w2, HH, lane, n0, k0);
      c0 = WMMA_BF16(a0, bf, c0);
      c1 = WMMA_BF16(a1, bf, c1);
      c2 = WMMA_BF16(a2, bf, c2);
    }
    int n = n0 + (lane & 15);
    float bias = b2[n];
    #pragma unroll
    for (int mt = 0; mt < 3; ++mt) {
      v8f cc = (mt == 0) ? c0 : ((mt == 1) ? c1 : c2);
      #pragma unroll
      for (int r = 0; r < 8; ++r) {
        int m = mt * 16 + (lane >> 4) * 8 + r;
        float x = cc[r] + bias + slots[(long)(rowBase + m) * HH + n];  // residual f32
        sX[m * SAW + n] = f2bf(x);
      }
    }
  }
  __syncthreads();

  // ---------------- LayerNorm per row, wave32 shuffle reduction --------------
  #pragma unroll 1
  for (int m = wave; m < MROWS; m += 8) {
    float s = 0.f, s2 = 0.f;
    #pragma unroll 1
    for (int i = 0; i < 32; ++i) {
      float x = (float)sX[m * SAW + i * 32 + lane];
      s += x; s2 += x * x;
    }
    #pragma unroll
    for (int off = 16; off > 0; off >>= 1) {
      s  += __shfl_xor(s,  off, 32);
      s2 += __shfl_xor(s2, off, 32);
    }
    float mu   = s * (1.f / HH);
    float var  = s2 * (1.f / HH) - mu * mu;
    float rstd = rsqrtf(var + 1e-5f);
    float* orow = out + (long)(rowBase + m) * HH;
    #pragma unroll 1
    for (int i = 0; i < 32; ++i) {
      int c = i * 32 + lane;
      float x = (float)sX[m * SAW + c];
      orow[c] = (x - mu) * rstd * gamma[c] + beta[c];
    }
  }
}

extern "C" void kernel_launch(void* const* d_in, const int* in_sizes, int n_in,
                              void* d_out, int out_size, void* d_ws, size_t ws_size,
                              hipStream_t stream) {
  (void)in_sizes; (void)n_in; (void)out_size; (void)d_ws; (void)ws_size;
  const float* slots = (const float*)d_in[0];
  const float* w1    = (const float*)d_in[1];
  const float* b1    = (const float*)d_in[2];
  const float* w2    = (const float*)d_in[3];
  const float* b2    = (const float*)d_in[4];
  const float* gamma = (const float*)d_in[5];
  const float* beta  = (const float*)d_in[6];
  float* out = (float*)d_out;

  (void)hipFuncSetAttribute(
      reinterpret_cast<const void*>(StructuredSlotCommunication_83562883711091_kernel),
      hipFuncAttributeMaxDynamicSharedMemorySize, SMEM_BYTES);

  dim3 grid(BB / NB), block(256);
  StructuredSlotCommunication_83562883711091_kernel<<<grid, block, SMEM_BYTES, stream>>>(
      slots, w1, b1, w2, b2, gamma, beta, out);
}
